// TokenPackerModule_48146583388486
// MI455X (gfx1250) — matrix operand order repaired and checked by reading, not compiled, not run on hardware
//
#include <hip/hip_runtime.h>
#include <math.h>

// ---------------------------------------------------------------------------
// TokenPacker attention block for MI455X (gfx1250), wave32 + WMMA f16.
//   t=5, hw=729, DIM=3456, E=1024, H=8, D=128, L=t*hw=3645
// The divide_feature permutation before attention is undone after out_proj;
// softmax attention is permutation-equivariant, so we compute in original
// token order throughout.
// ---------------------------------------------------------------------------

typedef __attribute__((ext_vector_type(16))) _Float16 v16h;
typedef __attribute__((ext_vector_type(8)))  float    v8f;
typedef _Float16 h16;

#define LN_EPS 1e-5f

__device__ __forceinline__ float gelu_exact(float x) {
    return 0.5f * x * (1.0f + erff(x * 0.70710678118654752440f));
}

struct U32B { uint4 lo, hi; };

// A-operand fragment (16x32, MxK): lane<16 holds K=[koff..koff+7],[koff+16..koff+23]
// with koff = 0 (lanes 0-15) / 8 (lanes 16-31), per CDNA5 ISA 16-bit A layout.
__device__ __forceinline__ v16h load_a_frag(const h16* p /* row base + koff */) {
    U32B u;
    u.lo = *(const uint4*)(p);
    u.hi = *(const uint4*)(p + 16);
    return __builtin_bit_cast(v16h, u);
}

// B-operand fragment (32x16, KxN): lane%16 = N; K=[koff..koff+7],[koff+16..koff+23], stride ld.
__device__ __forceinline__ v16h load_b_frag(const h16* p /* &B[k=koff][n0+lane%16] */, int ld) {
    v16h v;
#pragma unroll
    for (int s = 0; s < 8; ++s) { v[s] = p[s * ld]; v[s + 8] = p[(16 + s) * ld]; }
    return v;
}

// ---------------------------------------------------------------------------
// fp32 -> f16 conversion
// ---------------------------------------------------------------------------
__global__ void f32_to_f16_kernel(const float* __restrict__ in,
                                  h16* __restrict__ out, size_t n) {
    size_t i = (size_t)blockIdx.x * blockDim.x + threadIdx.x;
    size_t stride = (size_t)gridDim.x * blockDim.x;
    for (; i < n; i += stride) out[i] = (h16)in[i];
}

// ---------------------------------------------------------------------------
// Tiled WMMA GEMM:  C(M,N) = act( A(M,K) @ W(N,K)^T + bias ) [+ residual]
//   A,W in f16 row-major (K contiguous). BM=BN=128, BK=64.
//   256 threads = 8 waves; wave (wm 0..3, wn 0..1) computes 32x64.
// ---------------------------------------------------------------------------
template<int ACT /*0 none,1 gelu*/, bool OUT_F32, bool RES>
__global__ __launch_bounds__(256)
void gemm_wmma(const h16* __restrict__ A, const h16* __restrict__ W,
               const float* __restrict__ bias, const float* __restrict__ resid,
               h16* __restrict__ Ch, float* __restrict__ Cf,
               int M, int N, int K) {
    constexpr int BM = 128, BN = 128, BK = 64;
    constexpr int LDA = BK + 8;   // 72 halves = 144B, 16B aligned rows
    constexpr int LDB = BN + 8;   // 136 halves = 272B
    __shared__ __align__(16) h16 As[BM * LDA];
    __shared__ __align__(16) h16 Bs[BK * LDB];

    const int tid  = threadIdx.x;
    const int lane = tid & 31;
    const int wave = tid >> 5;
    const int wm   = wave >> 1;        // 0..3 -> 32-row slab
    const int wn   = wave & 1;         // 0..1 -> 64-col slab
    const int bM   = blockIdx.y * BM;
    const int bN   = blockIdx.x * BN;

    v8f acc[2][4];
#pragma unroll
    for (int i = 0; i < 2; ++i)
#pragma unroll
        for (int j = 0; j < 4; ++j)
#pragma unroll
            for (int r = 0; r < 8; ++r) acc[i][j][r] = 0.0f;

    const int lrow = tid >> 3;          // 0..31
    const int lcol = (tid & 7) * 8;     // 0..56 (halves)
    const int m_in = lane & 15;
    const int koff = (lane < 16) ? 0 : 8;

    for (int k0 = 0; k0 < K; k0 += BK) {
        // --- stage A tile (row-major) and W tile (K-transposed) into LDS ---
#pragma unroll
        for (int it = 0; it < 4; ++it) {
            int r  = lrow + it * 32;                 // 0..127
            int gm = bM + r;
            uint4 d = make_uint4(0u, 0u, 0u, 0u);
            if (gm < M) d = *(const uint4*)(A + (size_t)gm * K + k0 + lcol);
            *(uint4*)(&As[r * LDA + lcol]) = d;
        }
#pragma unroll
        for (int it = 0; it < 4; ++it) {
            int r  = lrow + it * 32;                 // N index in tile (N%128==0)
            int gn = bN + r;
            uint4 d = *(const uint4*)(W + (size_t)gn * K + k0 + lcol);
            const h16* hp = (const h16*)&d;
#pragma unroll
            for (int e = 0; e < 8; ++e) Bs[(lcol + e) * LDB + r] = hp[e];
        }
        if (k0 + BK < K) {   // speculative next-tile prefetch (global_prefetch_b8)
            int pa = bM + lrow; if (pa >= M) pa = M - 1;
            __builtin_prefetch(A + (size_t)pa * K + k0 + BK + lcol, 0, 1);
            __builtin_prefetch(W + (size_t)(bN + lrow) * K + k0 + BK + lcol, 0, 1);
        }
        __syncthreads();

        // --- 2 k-steps of 32 per LDS tile ---
#pragma unroll
        for (int ks = 0; ks < BK / 32; ++ks) {
            const int kb = ks * 32;
            v16h af[2];
#pragma unroll
            for (int i = 0; i < 2; ++i)
                af[i] = load_a_frag(&As[(wm * 32 + i * 16 + m_in) * LDA + kb + koff]);
            v16h bf[4];
#pragma unroll
            for (int j = 0; j < 4; ++j)
                bf[j] = load_b_frag(&Bs[(kb + koff) * LDB + wn * 64 + j * 16 + m_in], LDB);
#pragma unroll
            for (int i = 0; i < 2; ++i)
#pragma unroll
                for (int j = 0; j < 4; ++j)
                    acc[i][j] = __builtin_amdgcn_wmma_f32_16x16x32_f16(
                        false, af[i], false, bf[j], (short)0, acc[i][j], false, false);
        }
        __syncthreads();
    }

    // --- epilogue: C layout — VGPR r: lanes0-15 row r, lanes16-31 row r+8 ---
    const int rofs = (lane < 16) ? 0 : 8;
#pragma unroll
    for (int i = 0; i < 2; ++i)
#pragma unroll
        for (int j = 0; j < 4; ++j) {
            const int col = bN + wn * 64 + j * 16 + (lane & 15);
            const float bv = bias ? bias[col] : 0.0f;
#pragma unroll
            for (int r = 0; r < 8; ++r) {
                int row = bM + wm * 32 + i * 16 + r + rofs;
                if (row < M) {
                    float c = acc[i][j][r] + bv;
                    if (ACT == 1) c = gelu_exact(c);
                    if (RES) c += resid[(size_t)row * N + col];
                    if (OUT_F32) Cf[(size_t)row * N + col] = c;
                    else         Ch[(size_t)row * N + col] = (h16)c;
                }
            }
        }
}

// ---------------------------------------------------------------------------
// LayerNorm over last dim (E=1024): f32 in -> f16 out. One block per row.
// ---------------------------------------------------------------------------
__global__ __launch_bounds__(256)
void layernorm_kernel(const float* __restrict__ X, const float* __restrict__ g,
                      const float* __restrict__ b, h16* __restrict__ Y, int E) {
    const int row = blockIdx.x;
    const float* x = X + (size_t)row * E;
    float s = 0.f, s2 = 0.f;
    for (int i = threadIdx.x; i < E; i += blockDim.x) {
        float v = x[i]; s += v; s2 += v * v;
    }
    __shared__ float red[2][8];
    const int lane = threadIdx.x & 31, wv = threadIdx.x >> 5;
#pragma unroll
    for (int m = 16; m >= 1; m >>= 1) {
        s  += __shfl_xor(s,  m, 32);
        s2 += __shfl_xor(s2, m, 32);
    }
    if (lane == 0) { red[0][wv] = s; red[1][wv] = s2; }
    __syncthreads();
    if (wv == 0) {
        s  = (lane < 8) ? red[0][lane] : 0.f;
        s2 = (lane < 8) ? red[1][lane] : 0.f;
#pragma unroll
        for (int m = 4; m >= 1; m >>= 1) {
            s  += __shfl_xor(s,  m, 32);
            s2 += __shfl_xor(s2, m, 32);
        }
        if (lane == 0) { red[0][0] = s; red[1][0] = s2; }
    }
    __syncthreads();
    const float mean = red[0][0] / (float)E;
    const float var  = red[1][0] / (float)E - mean * mean;
    const float rstd = rsqrtf(var + LN_EPS);
    h16* y = Y + (size_t)row * E;
    for (int i = threadIdx.x; i < E; i += blockDim.x)
        y[i] = (h16)((x[i] - mean) * rstd * g[i] + b[i]);
}

// ---------------------------------------------------------------------------
// Flash attention, one head per blockIdx.y, 128 queries per block (8 waves x 16).
// Q/K/V: (L, 1024) f16, head h at columns [h*128, h*128+128). D=128, TK=64.
// ---------------------------------------------------------------------------
__global__ __launch_bounds__(256)
void flash_attn_wmma(const h16* __restrict__ Qg, const h16* __restrict__ Kg,
                     const h16* __restrict__ Vg, h16* __restrict__ Og,
                     int L, float scale) {
    constexpr int D = 128, TK = 64, E = 1024;
    constexpr int LKT = TK + 8;   // Kt[d][key]
    constexpr int LVT = D + 8;    // Vt[key][d]
    constexpr int LPW = TK + 8;   // Pw[wave][m][key]
    __shared__ __align__(16) h16 Kt[D * LKT];
    __shared__ __align__(16) h16 Vt[TK * LVT];
    __shared__ __align__(16) h16 Pw[8 * 16 * LPW];

    const int tid = threadIdx.x, lane = tid & 31, wave = tid >> 5;
    const int h = blockIdx.y;
    const int qbase = blockIdx.x * 128 + wave * 16;
    const int colbase = h * D;
    const int m_in = lane & 15;
    const int koff = (lane < 16) ? 0 : 8;
    const int keylane = lane & 15;

    // Q resident in A-fragments for the whole key loop
    int qrow = qbase + m_in; if (qrow >= L) qrow = L - 1;   // clamp (not stored)
    v16h qf[4];
#pragma unroll
    for (int kc = 0; kc < 4; ++kc)
        qf[kc] = load_a_frag(Qg + (size_t)qrow * E + colbase + kc * 32 + koff);

    v8f o[8];
#pragma unroll
    for (int j = 0; j < 8; ++j)
#pragma unroll
        for (int r = 0; r < 8; ++r) o[j][r] = 0.0f;
    float rowm[8], rowl[8];
#pragma unroll
    for (int r = 0; r < 8; ++r) { rowm[r] = -3.0e38f; rowl[r] = 0.0f; }

    const int trow = tid >> 4;          // 0..15
    const int tcol = (tid & 15) * 8;    // 0..120 halves

    for (int kt0 = 0; kt0 < L; kt0 += TK) {
        // ---- stage K (transposed) and V (row-major) tiles ----
#pragma unroll
        for (int it = 0; it < 4; ++it) {
            int r  = trow + it * 16;     // key in tile 0..63
            int gk = kt0 + r;
            uint4 dk = make_uint4(0u,0u,0u,0u), dv = make_uint4(0u,0u,0u,0u);
            if (gk < L) {
                dk = *(const uint4*)(Kg + (size_t)gk * E + colbase + tcol);
                dv = *(const uint4*)(Vg + (size_t)gk * E + colbase + tcol);
            }
            *(uint4*)(&Vt[r * LVT + tcol]) = dv;
            const h16* hp = (const h16*)&dk;
#pragma unroll
            for (int e = 0; e < 8; ++e) Kt[(tcol + e) * LKT + r] = hp[e];
        }
        __syncthreads();

        // ---- S = scale * Q K^T  (4 n-frags x 4 k-steps) ----
        v8f s[4];
#pragma unroll
        for (int j = 0; j < 4; ++j) {
#pragma unroll
            for (int r = 0; r < 8; ++r) s[j][r] = 0.0f;
#pragma unroll
            for (int kc = 0; kc < 4; ++kc) {
                v16h bf = load_b_frag(&Kt[(kc * 32 + koff) * LKT + j * 16 + m_in], LKT);
                s[j] = __builtin_amdgcn_wmma_f32_16x16x32_f16(
                    false, qf[kc], false, bf, (short)0, s[j], false, false);
            }
        }
        // scale + out-of-range key mask (value select, no divergence)
#pragma unroll
        for (int j = 0; j < 4; ++j) {
            const bool oob = (kt0 + j * 16 + keylane) >= L;
#pragma unroll
            for (int r = 0; r < 8; ++r) {
                float v = s[j][r] * scale;
                s[j][r] = oob ? -3.0e38f : v;
            }
        }
        // ---- online softmax: per-row (r, 16-lane group) max/sum ----
        float alpha[8];
#pragma unroll
        for (int r = 0; r < 8; ++r) {
            float mt = s[0][r];
#pragma unroll
            for (int j = 1; j < 4; ++j) mt = fmaxf(mt, s[j][r]);
#pragma unroll
            for (int m = 8; m >= 1; m >>= 1) mt = fmaxf(mt, __shfl_xor(mt, m, 32));
            float mnew = fmaxf(rowm[r], mt);
            alpha[r] = expf(rowm[r] - mnew);
            rowm[r] = mnew;
        }
#pragma unroll
        for (int r = 0; r < 8; ++r) {
            float ts = 0.0f;
#pragma unroll
            for (int j = 0; j < 4; ++j) {
                float p = expf(s[j][r] - rowm[r]);
                s[j][r] = p;
                ts += p;
            }
#pragma unroll
            for (int m = 8; m >= 1; m >>= 1) ts += __shfl_xor(ts, m, 32);
            rowl[r] = rowl[r] * alpha[r] + ts;
#pragma unroll
            for (int j = 0; j < 8; ++j) o[j][r] *= alpha[r];
        }
        // ---- re-fragment P (C layout -> A layout) through per-wave LDS ----
        h16* pw = &Pw[wave * 16 * LPW];
        {
            const int mrow = (lane < 16) ? 0 : 8;
#pragma unroll
            for (int j = 0; j < 4; ++j)
#pragma unroll
                for (int r = 0; r < 8; ++r)
                    pw[(r + mrow) * LPW + j * 16 + keylane] = (h16)s[j][r];
        }
        // ---- O += P V ----
#pragma unroll
        for (int kc = 0; kc < 2; ++kc) {
            v16h af = load_a_frag(&pw[m_in * LPW + kc * 32 + koff]);
#pragma unroll
            for (int j = 0; j < 8; ++j) {
                v16h bf = load_b_frag(&Vt[(kc * 32 + koff) * LVT + j * 16 + m_in], LVT);
                o[j] = __builtin_amdgcn_wmma_f32_16x16x32_f16(
                    false, af, false, bf, (short)0, o[j], false, false);
            }
        }
        __syncthreads();
    }

    // ---- epilogue: O /= rowsum, store f16 ----
    const int rofs = (lane < 16) ? 0 : 8;
#pragma unroll
    for (int r = 0; r < 8; ++r) {
        float inv = 1.0f / rowl[r];
        int row = qbase + r + rofs;
        if (row < L) {
#pragma unroll
            for (int j = 0; j < 8; ++j)
                Og[(size_t)row * E + colbase + j * 16 + keylane] = (h16)(o[j][r] * inv);
        }
    }
}

// ---------------------------------------------------------------------------
// Host-side orchestration
// ---------------------------------------------------------------------------
static inline void launch_cvt(const float* in, h16* out, size_t n, hipStream_t s) {
    size_t blocks = (n + 255) / 256; if (blocks > 4096) blocks = 4096;
    f32_to_f16_kernel<<<(unsigned)blocks, 256, 0, s>>>(in, out, n);
}

extern "C" void kernel_launch(void* const* d_in, const int* in_sizes, int n_in,
                              void* d_out, int out_size, void* d_ws, size_t ws_size,
                              hipStream_t stream) {
    constexpr int L = 5 * 729;      // 3645
    constexpr int DIM = 3456;
    constexpr int E = 1024;
    constexpr int H = 8;
    constexpr int D = 128;

    const float* x         = (const float*)d_in[0];
    const float* in_proj_w = (const float*)d_in[19];
    const float* in_proj_b = (const float*)d_in[20];
    const float* out_proj_w= (const float*)d_in[21];
    const float* out_proj_b= (const float*)d_in[22];
    const float* to_out_w  = (const float*)d_in[23];
    float* out = (float*)d_out;

    // ---- carve workspace (16B+ aligned chunks) ----
    char* cur = (char*)d_ws;
    auto carve = [&](size_t bytes) -> void* {
        void* p = (void*)cur;
        cur += (bytes + 255) & ~(size_t)255;
        return p;
    };
    h16*   xh      = (h16*)  carve((size_t)L * DIM * 2);
    h16*   w1h     = (h16*)  carve((size_t)E * DIM * 2);
    h16*   w2h     = (h16*)  carve((size_t)E * E * 2);
    h16*   hbuf    = (h16*)  carve((size_t)L * E * 2);
    float* mlpf    = (float*)carve((size_t)L * E * 4);
    h16*   ln3[3];
    ln3[0]         = (h16*)  carve((size_t)L * E * 2);
    ln3[1]         = (h16*)  carve((size_t)L * E * 2);
    ln3[2]         = (h16*)  carve((size_t)L * E * 2);
    h16*   inprojh = (h16*)  carve((size_t)3 * E * E * 2);
    h16*   qp      = (h16*)  carve((size_t)L * E * 2);
    h16*   kp      = (h16*)  carve((size_t)L * E * 2);
    h16*   vp      = (h16*)  carve((size_t)L * E * 2);
    h16*   attno   = (h16*)  carve((size_t)L * E * 2);
    h16*   outpw   = (h16*)  carve((size_t)E * E * 2);
    h16*   outp    = (h16*)  carve((size_t)L * E * 2);
    h16*   toow    = (h16*)  carve((size_t)DIM * E * 2);
    (void)ws_size; (void)n_in; (void)in_sizes; (void)out_size;

    // ---- precision conversions ----
    launch_cvt(x,          xh,      (size_t)L * DIM,   stream);
    launch_cvt(in_proj_w,  inprojh, (size_t)3 * E * E, stream);
    launch_cvt(out_proj_w, outpw,   (size_t)E * E,     stream);
    launch_cvt(to_out_w,   toow,    (size_t)DIM * E,   stream);

    const dim3 blk(256);
    const dim3 gE((E   / 128), (L + 127) / 128);   // N=1024
    const dim3 gD((DIM / 128), (L + 127) / 128);   // N=3456

    // ---- q/k/v MLP + LayerNorm ----
    for (int sIdx = 0; sIdx < 3; ++sIdx) {
        const int base = 1 + 6 * sIdx;
        const float* w1  = (const float*)d_in[base + 0];
        const float* b1  = (const float*)d_in[base + 1];
        const float* w2  = (const float*)d_in[base + 2];
        const float* b2  = (const float*)d_in[base + 3];
        const float* lng = (const float*)d_in[base + 4];
        const float* lnb = (const float*)d_in[base + 5];

        launch_cvt(w1, w1h, (size_t)E * DIM, stream);
        launch_cvt(w2, w2h, (size_t)E * E,   stream);

        // h = GELU(x @ w1^T + b1)
        gemm_wmma<1, false, false><<<gE, blk, 0, stream>>>(
            xh, w1h, b1, nullptr, hbuf, nullptr, L, E, DIM);
        // m = h @ w2^T + b2  (f32 for LN)
        gemm_wmma<0, true, false><<<gE, blk, 0, stream>>>(
            hbuf, w2h, b2, nullptr, nullptr, mlpf, L, E, E);
        // LN -> f16
        layernorm_kernel<<<L, blk, 0, stream>>>(mlpf, lng, lnb, ln3[sIdx], E);
    }

    // ---- packed in_proj ----
    gemm_wmma<0, false, false><<<gE, blk, 0, stream>>>(
        ln3[0], inprojh,                      in_proj_b,         nullptr, qp, nullptr, L, E, E);
    gemm_wmma<0, false, false><<<gE, blk, 0, stream>>>(
        ln3[1], inprojh + (size_t)E * E,      in_proj_b + E,     nullptr, kp, nullptr, L, E, E);
    gemm_wmma<0, false, false><<<gE, blk, 0, stream>>>(
        ln3[2], inprojh + (size_t)2 * E * E,  in_proj_b + 2 * E, nullptr, vp, nullptr, L, E, E);

    // ---- flash attention (8 heads) ----
    {
        dim3 gA((L + 127) / 128, H);
        float scale = 1.0f / sqrtf((float)D);
        flash_attn_wmma<<<gA, blk, 0, stream>>>(qp, kp, vp, attno, L, scale);
    }

    // ---- out_proj ----
    gemm_wmma<0, false, false><<<gE, blk, 0, stream>>>(
        attno, outpw, out_proj_b, nullptr, outp, nullptr, L, E, E);

    // ---- to_out + residual: d_out = x + outp @ to_out_w^T ----
    gemm_wmma<0, true, true><<<gD, blk, 0, stream>>>(
        outp, toow, nullptr, x, nullptr, out, L, DIM, E);
}
